// CompressedFP8Linear_90357521973484
// MI455X (gfx1250) — matrix operand order repaired and checked by reading, not compiled, not run on hardware
//
#include <hip/hip_runtime.h>

// ---- problem constants (B=4, S=32 -> M=128; IN=OUT=8192) ----
#define IN_DIM  8192
#define OUT_DIM 8192
#define M_DIM   128
#define KC      128          // K-chunk staged in LDS
#define NT      32           // N-tile per workgroup -> 256 workgroups
#define LDSROW  (KC + 8)     // pad 8 halfs (16B): row stride 272B, 16B aligned

#define XLD ((M_DIM * KC) / (4 * 256))   // 16 float4 per thread (x chunk)
#define WLD ((NT * KC) / (4 * 256))      // 4  float4 per thread (w chunk)

typedef __attribute__((ext_vector_type(16))) _Float16 v16h;
typedef __attribute__((ext_vector_type(8)))  _Float16 v8h;
typedef __attribute__((ext_vector_type(4)))  _Float16 v4h;
typedef __attribute__((ext_vector_type(8)))  float    v8f;
typedef __attribute__((ext_vector_type(4)))  float    v4f;   // clang vector: OK for nontemporal builtins

__global__ __launch_bounds__(256)
void fp8lin_wmma_kernel(const float* __restrict__ x,      // [128, 8192] fp32
                        const float* __restrict__ w,      // [8192, 8192] fp32 (fp8-representable)
                        const float* __restrict__ scale,  // [8192] fp32, per OUT channel
                        const _Float16* __restrict__ bias,// [8192] fp16
                        float* __restrict__ out)          // [128, 8192] fp32
{
    __shared__ _Float16 sA[M_DIM][LDSROW];   // x chunk,  f16  (~34 KB)
    __shared__ _Float16 sB[NT][LDSROW];      // w chunk,  f16  (~8.5 KB)

    const int tid   = threadIdx.x;
    const int wave  = tid >> 5;
    const int lane  = tid & 31;
    const int lhalf = lane >> 4;   // 0: lanes 0-15, 1: lanes 16-31
    const int l16   = lane & 15;

    const int nBase = blockIdx.x * NT;   // 8192/32 = 256 blocks
    const int mBase = wave * 16;         // 8 waves cover M=128

    v8f acc[2];
    #pragma unroll
    for (int t = 0; t < 2; ++t) acc[t] = (v8f){0.f,0.f,0.f,0.f,0.f,0.f,0.f,0.f};

    const v4f* x4 = (const v4f*)x;
    const v4f* w4 = (const v4f*)w;

    // register prefetch buffers (chunk i+1 loads overlap chunk i WMMAs)
    v4f xr[XLD];
    v4f wr[WLD];

    // ---- prologue: load chunk 0 ----
    #pragma unroll
    for (int i = 0; i < XLD; ++i) {
        int g = tid + 256 * i;
        xr[i] = x4[(g >> 5) * (IN_DIM / 4) + (g & 31)];        // x: cached (L2-resident)
    }
    #pragma unroll
    for (int i = 0; i < WLD; ++i) {
        int g = tid + 256 * i;
        wr[i] = __builtin_nontemporal_load(
            &w4[(size_t)(nBase + (g >> 5)) * (IN_DIM / 4) + (g & 31)]); // w: one-shot stream
    }

    for (int k0 = 0; k0 < IN_DIM; k0 += KC) {
        __syncthreads();   // previous chunk's LDS reads done

        // ---- convert + store staged registers into LDS ----
        #pragma unroll
        for (int i = 0; i < XLD; ++i) {
            int g = tid + 256 * i;
            v4h h;
            h[0] = (_Float16)xr[i].x; h[1] = (_Float16)xr[i].y;
            h[2] = (_Float16)xr[i].z; h[3] = (_Float16)xr[i].w;
            *(v4h*)&sA[g >> 5][(g & 31) * 4] = h;
        }
        #pragma unroll
        for (int i = 0; i < WLD; ++i) {
            int g = tid + 256 * i;
            v4h h;
            h[0] = (_Float16)wr[i].x; h[1] = (_Float16)wr[i].y;
            h[2] = (_Float16)wr[i].z; h[3] = (_Float16)wr[i].w;
            *(v4h*)&sB[g >> 5][(g & 31) * 4] = h;
        }
        __syncthreads();

        // ---- prefetch next chunk into registers (overlaps WMMA below) ----
        const int kn = k0 + KC;
        if (kn < IN_DIM) {
            #pragma unroll
            for (int i = 0; i < XLD; ++i) {
                int g = tid + 256 * i;
                xr[i] = x4[(g >> 5) * (IN_DIM / 4) + (kn >> 2) + (g & 31)];
            }
            #pragma unroll
            for (int i = 0; i < WLD; ++i) {
                int g = tid + 256 * i;
                wr[i] = __builtin_nontemporal_load(
                    &w4[(size_t)(nBase + (g >> 5)) * (IN_DIM / 4) + (kn >> 2) + (g & 31)]);
            }
        }

        // ---- compute: 4 K-steps of 32, 2 N-subtiles per wave ----
        const _Float16* aRow = &sA[mBase + l16][0];
        #pragma unroll
        for (int kk = 0; kk < KC; kk += 32) {
            // A frag 16x32 f16: lanes0-15 K 0-15, lanes16-31 K 16-31
            union { v16h v; v8h h[2]; } af;
            af.h[0] = *(const v8h*)(aRow + kk + lhalf * 16);
            af.h[1] = *(const v8h*)(aRow + kk + lhalf * 16 + 8);

            const int kb = kk + lhalf * 16;  // B frag 32x16: lanes0-15 K 0-15, lanes16-31 K 16-31
            #pragma unroll
            for (int t = 0; t < 2; ++t) {
                union { v16h v; v8h h[2]; } bf;
                const _Float16* bRow = &sB[t * 16 + l16][0];
                bf.h[0] = *(const v8h*)(bRow + kb);
                bf.h[1] = *(const v8h*)(bRow + kb + 8);
                acc[t] = __builtin_amdgcn_wmma_f32_16x16x32_f16(
                    /*neg_a=*/false, af.v, /*neg_b=*/false, bf.v,
                    /*c_mod=*/(short)0, acc[t], /*reuse_a=*/false, /*reuse_b=*/false);
            }
        }
    }

    // ---- epilogue: out = acc * scale[n] + bias[n]  (fp32, non-temporal stores) ----
    // C/D layout: VGPR r, lanes 0-15 -> row M=r, lanes 16-31 -> row M=8+r; col N = lane%16
    #pragma unroll
    for (int t = 0; t < 2; ++t) {
        const int ncol = nBase + t * 16 + l16;
        const float sc = scale[ncol];
        const float bv = (float)bias[ncol];
        #pragma unroll
        for (int r = 0; r < 8; ++r) {
            const int m = mBase + r + 8 * lhalf;
            __builtin_nontemporal_store(acc[t][r] * sc + bv,
                                        &out[(size_t)m * OUT_DIM + ncol]);
        }
    }
}

extern "C" void kernel_launch(void* const* d_in, const int* in_sizes, int n_in,
                              void* d_out, int out_size, void* d_ws, size_t ws_size,
                              hipStream_t stream) {
    (void)in_sizes; (void)n_in; (void)out_size; (void)d_ws; (void)ws_size;
    const float*    x     = (const float*)d_in[0];
    const float*    w     = (const float*)d_in[1];
    const float*    scale = (const float*)d_in[2];
    const _Float16* bias  = (const _Float16*)d_in[3];
    float*          out   = (float*)d_out;

    dim3 grid(OUT_DIM / NT);   // 256 workgroups
    dim3 block(256);           // 8 wave32
    fp8lin_wmma_kernel<<<grid, block, 0, stream>>>(x, w, scale, bias, out);
}
